// ManyDeepFeatureMatcher_52785148067938
// MI455X (gfx1250) — compile-verified, matrix operand order checked by reading, not run on hardware
//
#include <hip/hip_runtime.h>
#include <hip/hip_bf16.h>
#include <math.h>

// ---------------------------------------------------------------------------
// ManyDeepFeatureMatcher for MI455X (gfx1250, wave32, WMMA bf16 16x16x32)
//
// Fused GEMM + running top-2; sim matrix never materialized.
// v3: block = 8 row-bands sharing one column split; B panels staged through
// LDS (double-buffered, 1 barrier/tile) -> 8x less L2 traffic (128 FLOP/B),
// tile t+1 global loads overlap tile t WMMA chain; pure-select top-2 (no
// canonicalize ops).
// ---------------------------------------------------------------------------

typedef __attribute__((ext_vector_type(16))) __bf16          v16bf;
typedef __attribute__((ext_vector_type(8)))  __bf16          v8bf;
typedef __attribute__((ext_vector_type(8)))  float           v8f;
typedef __attribute__((ext_vector_type(8)))  unsigned short  v8us;

#define NDESC 8192
#define NCH   256
#define NSPLIT 4
#define COLS_PER_SPLIT (NDESC / NSPLIT)        // 2048
#define TILES_PER_SPLIT (COLS_PER_SPLIT / 16)  // 128
#define LDS_STRIDE 264   // halves per descriptor row (528B = 132 DW = 4 mod 64 banks)
#define RATIO 0.9f
#define EPS 1e-8f

// float -> bf16 (round-to-nearest-even), bit-level.
__device__ __forceinline__ unsigned short f32_to_bf16(float v) {
    unsigned int x = __float_as_uint(v);
    x += 0x7FFFu + ((x >> 16) & 1u);
    return (unsigned short)(x >> 16);
}

// ---------------------------------------------------------------------------
// Kernel 1: column L2-normalize + transpose to row-major [desc][k] bf16.
// ---------------------------------------------------------------------------
__global__ __launch_bounds__(256)
void norm_kernel(const float* __restrict__ A, const float* __restrict__ B,
                 unsigned short* __restrict__ dA, unsigned short* __restrict__ dB) {
    int tid = blockIdx.x * blockDim.x + threadIdx.x;
    int j = tid & (NDESC - 1);
    const float* src = (tid < NDESC) ? A : B;
    unsigned short* dst = (tid < NDESC) ? dA : dB;

    float s = 0.0f;
#pragma unroll 8
    for (int c = 0; c < NCH; ++c) {
        float v = src[(size_t)c * NDESC + j];
        s += v * v;
    }
    float inv = 1.0f / sqrtf(s);

    for (int c = 0; c < NCH; c += 8) {
        v8us pk;
#pragma unroll
        for (int u = 0; u < 8; ++u)
            pk[u] = f32_to_bf16(src[(size_t)(c + u) * NDESC + j] * inv);
        *(v8us*)(dst + (size_t)j * NCH + c) = pk;
    }
}

// ---------------------------------------------------------------------------
// Kernel 2: fused sim-GEMM + running top-2 over columns.
//   sim[i][j] = dot(Rows[i][:], Cols[j][:])  (row-major [desc][256] bf16)
//
// Block = 8 waves = 8 row-bands (128 rows), all sharing one column split of
// 2048 columns. Per 16-column tile the block cooperatively stages the
// 16x256 bf16 B panel into LDS (8KB, double-buffered, padded stride).
// Each wave holds its 16x256 A panel in 64 VGPRs and runs 8 back-to-back
// bf16 WMMAs per tile out of LDS fragments.
//
// 16-bit WMMA fragment layout (ISA 7.12.2): lane L -> m/n = L&15, h = L>>4;
// VGPR0-3 = K in [8h,8h+8), VGPR4-7 = K in [16+8h,16+8h+8).
// ---------------------------------------------------------------------------
__global__ __launch_bounds__(256)
void simtop2_kernel(const unsigned short* __restrict__ Rows,
                    const unsigned short* __restrict__ Cols,
                    float4* __restrict__ part) {
    __shared__ unsigned short bbuf[2][16 * LDS_STRIDE];   // 2 x 8.25KB

    const int tid   = threadIdx.x;
    const int lane  = tid & 31;
    const int wave  = tid >> 5;
    const int h     = lane >> 4;
    const int n     = lane & 15;
    const int split = blockIdx.x & (NSPLIT - 1);
    const int row0  = (blockIdx.x >> 2) * 128 + wave * 16;
    const int colBase = split * COLS_PER_SPLIT;

    // Staging role: thread copies 32B chunk sc of descriptor sd each tile.
    const int sd = tid >> 4;    // 0..15
    const int sc = tid & 15;    // 0..15

    // A panel: 8 k-chunk fragments, loaded once from global.
    const unsigned short* aBase = Rows + (size_t)(row0 + n) * NCH + h * 8;
    v16bf afrag[8];
#pragma unroll
    for (int kc = 0; kc < 8; ++kc) {
        v8bf lo = *(const v8bf*)(aBase + kc * 32);
        v8bf hi = *(const v8bf*)(aBase + kc * 32 + 16);
        afrag[kc] = __builtin_shufflevector(lo, hi,
                        0,1,2,3,4,5,6,7, 8,9,10,11,12,13,14,15);
    }

    // Running top-2 per lane for the 8 rows (M = 8h+v) this lane owns in D.
    float t1[8], t2[8]; int i1[8];
#pragma unroll
    for (int v = 0; v < 8; ++v) { t1[v] = -3.0f; t2[v] = -3.0f; i1[v] = 0; }

    // Prologue: stage tile 0 into buffer 0.
    {
        const unsigned short* g = Cols + (size_t)(colBase + sd) * NCH + sc * 16;
        v8us r0 = *(const v8us*)(g);
        v8us r1 = *(const v8us*)(g + 8);
        unsigned short* l = &bbuf[0][sd * LDS_STRIDE + sc * 16];
        *(v8us*)(l)     = r0;
        *(v8us*)(l + 8) = r1;
    }
    __syncthreads();

    for (int t = 0; t < TILES_PER_SPLIT; ++t) {
        const int cur = t & 1;
        const int nxt = cur ^ 1;
        const int col0 = colBase + t * 16;
        const bool have = (t + 1 < TILES_PER_SPLIT);

        // Issue next tile's global loads BEFORE the WMMA chain (latency overlap).
        v8us r0 = {}, r1 = {};
        if (have) {
            const unsigned short* g =
                Cols + (size_t)(col0 + 16 + sd) * NCH + sc * 16;
            __builtin_prefetch(g + 16 * NCH, 0, 0);    // global_prefetch_b8 (t+2)
            r0 = *(const v8us*)(g);
            r1 = *(const v8us*)(g + 8);
        }

        // Compute tile t from LDS fragments: 16 ds_load_b128 + 8 WMMAs.
        v8f acc = {};
#pragma unroll
        for (int kc = 0; kc < 8; ++kc) {
            const unsigned short* fb =
                &bbuf[cur][n * LDS_STRIDE + kc * 32 + h * 8];
            v8bf lo = *(const v8bf*)(fb);
            v8bf hi = *(const v8bf*)(fb + 16);
            v16bf bfrag = __builtin_shufflevector(lo, hi,
                              0,1,2,3,4,5,6,7, 8,9,10,11,12,13,14,15);
            acc = __builtin_amdgcn_wmma_f32_16x16x32_bf16(
                      false, afrag[kc], false, bfrag,
                      (short)0, acc, false, false);
        }

        // Branchless running top-2 (compare + select only; no canonicalize).
        const int col = col0 + n;
#pragma unroll
        for (int v = 0; v < 8; ++v) {
            float val   = acc[v];                 // sim[row0+8h+v][col]
            bool  gt    = val > t1[v];
            float small = gt ? t1[v] : val;
            i1[v] = gt ? col : i1[v];
            t1[v] = gt ? val : t1[v];
            t2[v] = (small > t2[v]) ? small : t2[v];
        }

        // Stage tile t+1 into the other LDS buffer.
        if (have) {
            unsigned short* l = &bbuf[nxt][sd * LDS_STRIDE + sc * 16];
            *(v8us*)(l)     = r0;
            *(v8us*)(l + 8) = r1;
        }
        __syncthreads();
    }

    // Branchless top-2 merge across the 16 lanes of each half.
#pragma unroll
    for (int off = 1; off < 16; off <<= 1) {
#pragma unroll
        for (int v = 0; v < 8; ++v) {
            float o1 = __shfl_xor(t1[v], off, 32);
            float o2 = __shfl_xor(t2[v], off, 32);
            int   oi = __shfl_xor(i1[v], off, 32);
            bool  gt    = o1 > t1[v];
            float small = gt ? t1[v] : o1;        // loser's best
            float w2    = gt ? o2 : t2[v];        // winner's second
            i1[v] = gt ? oi : i1[v];
            t1[v] = gt ? o1 : t1[v];
            t2[v] = (small > w2) ? small : w2;
        }
    }

    if (n == 0) {   // lanes 0 (rows 0-7) and 16 (rows 8-15) write partials
#pragma unroll
        for (int v = 0; v < 8; ++v) {
            int row = row0 + 8 * h + v;
            part[(size_t)row * NSPLIT + split] =
                make_float4(t1[v], t2[v], __int_as_float(i1[v]), 0.0f);
        }
    }
}

// ---------------------------------------------------------------------------
// Kernel 3: merge NSPLIT partial top-2s per row; emit nn, ratio, top-sim.
// ---------------------------------------------------------------------------
__global__ __launch_bounds__(256)
void reduce_kernel(const float4* __restrict__ partA, const float4* __restrict__ partB,
                   int* __restrict__ nnA, float* __restrict__ ratioA,
                   float* __restrict__ simA,
                   int* __restrict__ nnB, float* __restrict__ ratioB) {
    int tid = blockIdx.x * blockDim.x + threadIdx.x;
    int row = tid & (NDESC - 1);
    const float4* p = ((tid < NDESC) ? partA : partB) + (size_t)row * NSPLIT;

    float t1 = -3.0f, t2 = -3.0f; int i1 = 0;
#pragma unroll
    for (int s = 0; s < NSPLIT; ++s) {
        float4 q = p[s];
        bool  gt    = q.x > t1;
        float small = gt ? t1 : q.x;
        float w2    = gt ? q.y : t2;
        i1 = gt ? __float_as_int(q.z) : i1;
        t1 = gt ? q.x : t1;
        t2 = (small > w2) ? small : w2;
    }
    float d1 = 2.0f - 2.0f * t1;
    float d2 = 2.0f - 2.0f * t2;
    float r  = d1 / (d2 + EPS);
    if (tid < NDESC) { nnA[row] = i1; ratioA[row] = r; simA[row] = t1; }
    else             { nnB[row] = i1; ratioB[row] = r; }
}

// ---------------------------------------------------------------------------
// Kernel 4: mutual-NN + ratio mask; write all 32768 output floats.
//   out[0 .. 16383]      matches (i, nn12[i]) or (-1,-1)
//   out[16384 .. 24575]  match_sim
//   out[24576 .. 32767]  mask (1.0 / 0.0)
// ---------------------------------------------------------------------------
__global__ __launch_bounds__(256)
void final_kernel(const int* __restrict__ nnA, const float* __restrict__ ratioA,
                  const float* __restrict__ simA,
                  const int* __restrict__ nnB, const float* __restrict__ ratioB,
                  float* __restrict__ out) {
    int i = blockIdx.x * blockDim.x + threadIdx.x;
    if (i >= NDESC) return;
    int j = nnA[i];
    bool m = (nnB[j] == i) && (ratioA[i] <= RATIO) && (ratioB[j] <= RATIO);
    out[2 * i]             = m ? (float)i : -1.0f;
    out[2 * i + 1]         = m ? (float)j : -1.0f;
    out[2 * NDESC + i]     = m ? simA[i]  : 0.0f;
    out[3 * NDESC + i]     = m ? 1.0f     : 0.0f;
}

// ---------------------------------------------------------------------------
// Workspace layout (~10.2 MB):
//   [0,   4MB)  dA  bf16 [8192][256]
//   [4MB, 8MB)  dB  bf16 [8192][256]
//   [8MB, 9MB)  partA float4 [8192][4]   (512KB used)
//   [9MB,10MB)  partB float4 [8192][4]   (512KB used)
//   [10MB, +160KB) nnA/ratioA/simA/nnB/ratioB
// ---------------------------------------------------------------------------
extern "C" void kernel_launch(void* const* d_in, const int* in_sizes, int n_in,
                              void* d_out, int out_size, void* d_ws, size_t ws_size,
                              hipStream_t stream) {
    (void)in_sizes; (void)n_in; (void)out_size; (void)ws_size;
    const float* A = (const float*)d_in[0];
    const float* B = (const float*)d_in[1];
    float* out = (float*)d_out;

    char* w = (char*)d_ws;
    unsigned short* dA = (unsigned short*)(w);
    unsigned short* dB = (unsigned short*)(w + (4u << 20));
    float4* pA = (float4*)(w + (8u << 20));
    float4* pB = (float4*)(w + (9u << 20));
    char* misc = w + (10u << 20);
    int*   nnA    = (int*)  (misc);
    float* ratioA = (float*)(misc + 32768);
    float* simA   = (float*)(misc + 65536);
    int*   nnB    = (int*)  (misc + 98304);
    float* ratioB = (float*)(misc + 131072);

    // 1) normalize + bf16 transpose (both matrices)
    norm_kernel<<<(2 * NDESC) / 256, 256, 0, stream>>>(A, B, dA, dB);

    // 2) fused GEMM + top-2, both directions
    //    grid = 64 row-groups x 4 splits = 256 blocks of 8 waves
    simtop2_kernel<<<(NDESC / 128) * NSPLIT, 256, 0, stream>>>(dA, dB, pA);
    simtop2_kernel<<<(NDESC / 128) * NSPLIT, 256, 0, stream>>>(dB, dA, pB);

    // 3) merge partials -> nn / ratio / sim
    reduce_kernel<<<(2 * NDESC) / 256, 256, 0, stream>>>(pA, pB, nnA, ratioA, simA, nnB, ratioB);

    // 4) mutual check + outputs
    final_kernel<<<NDESC / 256, 256, 0, stream>>>(nnA, ratioA, simA, nnB, ratioB, out);
}
